// CustomConv1d_76063870812463
// MI455X (gfx1250) — matrix operand order repaired and checked by reading, not compiled, not run on hardware
//
#include <hip/hip_runtime.h>

// Conv1d: x[32,256,4096] (f32) * w[256,256,5] (f32) + bias -> out[32,256,4096] (f32)
// Implicit-GEMM with bf16 WMMA (f32 accumulate), LDS-staged tiles.
//   M = C_out (64/block), N = W (128/block), K = C_in*5 (32-ci x 5-tap rounds).
// v3: use __builtin_amdgcn_sched_group_barrier to software-pipeline the inner
//     loop (DS loads for fragment n+1 issued before WMMA n) so the automatic
//     s_wait_dscnt relaxes from 0 to ~3 and LDS latency overlaps WMMA execution.

#define IN_CH  256
#define OUT_CH 256
#define BATCH  32
#define WIDTH  4096
#define KW     5
#define PADW   2

// sched_group_barrier masks
#define SGB_WMMA    0x008   // MFMA/WMMA group
#define SGB_DSREAD  0x100   // DS read group

typedef __bf16        v16bf __attribute__((ext_vector_type(16)));
typedef float         v8f   __attribute__((ext_vector_type(8)));
typedef unsigned int  v4u   __attribute__((ext_vector_type(4)));
typedef unsigned int  v8u   __attribute__((ext_vector_type(8)));

// f32 -> bf16 bits, round-to-nearest-even
__device__ __forceinline__ unsigned short f2bf(float f) {
    unsigned int u = __builtin_bit_cast(unsigned int, f);
    u += 0x7FFFu + ((u >> 16) & 1u);
    return (unsigned short)(u >> 16);
}

__global__ __launch_bounds__(256)
void conv1d_bf16_wmma(const float* __restrict__ x,
                      const float* __restrict__ wgt,
                      const float* __restrict__ bias,
                      float* __restrict__ out) {
    // Weights: [tap][co(64)][ci(32)+pad2]  -> row stride 68B = 17 banks (conflict-free A gather)
    __shared__ __align__(16) unsigned short sW[KW][64][34];
    // Input:   [w(132)][ci(32)+pad8]       -> row stride 80B, 16B-aligned for b128 B loads
    __shared__ __align__(16) unsigned short sX[132][40];

    const int tid   = threadIdx.x;
    const int lane  = tid & 31;
    const int wave  = tid >> 5;
    const int mtile = wave & 3;          // 4 waves along M (4 x 16 = 64 co)
    const int ntb   = (wave >> 2) * 64;  // 2 waves along N, each owns 64 w (4 subtiles of 16)
    const int h     = lane >> 4;         // lane half (K-half selector in fragments)
    const int nl    = lane & 15;         // M row (A) / N column (B) within tile

    const int wbase  = blockIdx.x * 128;
    const int cobase = blockIdx.y * 64;
    const int b      = blockIdx.z;

    v8f acc[4] = {};   // 4 N-subtiles of 16x16 f32

    for (int cb = 0; cb < 8; ++cb) {
        const int cibase = cb * 32;
        __syncthreads();   // protect previous round's LDS reads

        // ---- stage weights: w[cobase+r][cibase+c][t] -> sW[t][r][c] (bf16) ----
        // global is contiguous over (c,t): 160 elems per row r -> coalesced
        for (int idx = tid; idx < 64 * 160; idx += 256) {
            int r     = idx / 160;
            int inner = idx - r * 160;
            int c     = inner / 5;
            int t     = inner - c * 5;
            float val = wgt[(size_t)(cobase + r) * (IN_CH * KW)
                            + (size_t)(cibase + c) * KW + t];
            sW[t][r][c] = f2bf(val);
        }

        // ---- stage x slab with halo: 32 ci x 132 w (covers all 5 taps) ----
        for (int idx = tid; idx < 32 * 132; idx += 256) {
            int c  = idx / 132;
            int w  = idx - c * 132;          // consecutive threads -> consecutive w (coalesced)
            int wg = wbase - PADW + w;
            float val = 0.0f;
            if (wg >= 0 && wg < WIDTH)
                val = x[((size_t)b * IN_CH + (cibase + c)) * (size_t)WIDTH + wg];
            sX[w][c] = f2bf(val);
        }
        __syncthreads();

#pragma unroll
        for (int t = 0; t < KW; ++t) {
            // A fragment (16x32 bf16, contiguous K order, same as B): lane(h,nl): M=nl,
            //   pair p -> K = 2*p + 16*h  (two bf16 per b32 load)
            const unsigned short* wrow = &sW[t][mtile * 16 + nl][0];
            v8u araw;
#pragma unroll
            for (int p = 0; p < 8; ++p) {
                int k = 2 * p + 16 * h;
                araw[p] = *(const unsigned int*)(wrow + k);
            }
            v16bf afrag = __builtin_bit_cast(v16bf, araw);

#pragma unroll
            for (int nt = 0; nt < 4; ++nt) {
                // B fragment (contiguous K order, same as A): lane(h,nl) = column N=nl, K = e + 16*h
                // -> 16 contiguous ci at w-row (n_abs + t); two 16B LDS loads
                const unsigned short* brow = &sX[ntb + nt * 16 + nl + t][16 * h];
                v4u b0 = *(const v4u*)(brow);
                v4u b1 = *(const v4u*)(brow + 8);
                v8u braw;
#pragma unroll
                for (int i = 0; i < 4; ++i) { braw[i] = b0[i]; braw[i + 4] = b1[i]; }
                v16bf bfrag = __builtin_bit_cast(v16bf, braw);

                acc[nt] = __builtin_amdgcn_wmma_f32_16x16x32_bf16(
                    false, afrag, false, bfrag, (short)0, acc[nt], false, false);
            }
        }

        // ---- scheduling pipeline for this region ----
        // Per ci-round: 60 DS-read instructions (5 taps x 4 ds_load_2addr for A,
        // 20 fragments x 2 ds_load_b128 for B) feeding 20 WMMAs.
        // Enforce: 6 DS up front (A0 + B00), then 19 x (3 DS ahead, 1 WMMA),
        // then drain the last WMMA -> each WMMA has ~3 newer DS loads in flight
        // (s_wait_dscnt ~3 instead of 0).
        __builtin_amdgcn_sched_group_barrier(SGB_DSREAD, 6, 0);
#pragma unroll
        for (int i = 0; i < 19; ++i) {
            __builtin_amdgcn_sched_group_barrier(SGB_DSREAD, 3, 0);
            __builtin_amdgcn_sched_group_barrier(SGB_WMMA, 1, 0);
        }
        __builtin_amdgcn_sched_group_barrier(SGB_WMMA, 1, 0);
    }

    // ---- epilogue: bias + store (C/D layout: M = vgpr + 8*h, N = nl) ----
#pragma unroll
    for (int e = 0; e < 8; ++e) {
        int m  = e + 8 * h;
        int co = cobase + mtile * 16 + m;
        float bv = bias[co];
        size_t ob = ((size_t)b * OUT_CH + co) * (size_t)WIDTH + wbase;
#pragma unroll
        for (int nt = 0; nt < 4; ++nt) {
            out[ob + ntb + nt * 16 + nl] = acc[nt][e] + bv;
        }
    }
}

extern "C" void kernel_launch(void* const* d_in, const int* in_sizes, int n_in,
                              void* d_out, int out_size, void* d_ws, size_t ws_size,
                              hipStream_t stream) {
    const float* x    = (const float*)d_in[0];
    const float* wgt  = (const float*)d_in[1];
    const float* bias = (const float*)d_in[2];
    float* out        = (float*)d_out;

    dim3 grid(WIDTH / 128, OUT_CH / 64, BATCH);  // (32, 4, 32)
    dim3 block(256);                             // 8 wave32 waves
    conv1d_bf16_wmma<<<grid, block, 0, stream>>>(x, wgt, bias, out);
}
